// QuantizedLinear158_14654428414831
// MI455X (gfx1250) — matrix operand order repaired and checked by reading, not compiled, not run on hardware
//
#include <hip/hip_runtime.h>
#include <stdint.h>

typedef __attribute__((ext_vector_type(16))) _Float16    v16h;
typedef __attribute__((ext_vector_type(8)))  _Float16    v8h;
typedef __attribute__((ext_vector_type(8)))  float       v8f;
typedef __attribute__((ext_vector_type(4)))  float       v4f;
typedef __attribute__((ext_vector_type(16))) signed char v16c;

#define D_IN   4096
#define D_OUT  11008
#define M_TOT  8192          // B*S = 4*2048
#define BM     128
#define BN     128
#define BK     32
#define KTILES (D_IN / BK)   // 128
#define LDSS   40            // halves per LDS row: 32 + 8 pad (16B aligned, conflict-free)
#define WS_NEED ((size_t)D_OUT * (size_t)D_IN * sizeof(_Float16))  // 90,177,536 B

// ---------------- CDNA5 async global->LDS helpers (inline asm, ASYNCcnt) ---------------
__device__ __forceinline__ unsigned lds_addr32(const void* p) {
    // generic -> LDS addrspace -> 32-bit LDS byte offset
    return (unsigned)(uintptr_t)(__attribute__((address_space(3))) const void*)p;
}
__device__ __forceinline__ void async_cp_b128(const void* g, void* l) {
    // VDST = per-lane LDS byte address, VADDR = 64-bit global address
    asm volatile("global_load_async_to_lds_b128 %0, %1, off"
                 :: "v"(lds_addr32(l)), "v"(g) : "memory");
}
__device__ __forceinline__ void wait_async0() {
    asm volatile("s_wait_asynccnt 0x0" ::: "memory");
}

// ---------------- one-time W: int8 ternary -> f16 panel in workspace -------------------
__global__ __launch_bounds__(256)
void w_to_f16(const signed char* __restrict__ w, _Float16* __restrict__ wf) {
    const size_t base = ((size_t)blockIdx.x * 256 + threadIdx.x) * 16;
    v16c b = *(const v16c*)(w + base);
    v8h h0, h1;
#pragma unroll
    for (int i = 0; i < 8; ++i) {
        h0[i] = (_Float16)(float)b[i];
        h1[i] = (_Float16)(float)b[8 + i];
    }
    *(v8h*)(wf + base)     = h0;
    *(v8h*)(wf + base + 8) = h1;
}

// ---------------- stage x: global f32 -> regs -> f16 LDS -------------------------------
__device__ __forceinline__ void stageA_regs(const float* __restrict__ xp, int k0, v4f xr[4]) {
    const v4f* xv = (const v4f*)(xp + k0);
    xr[0] = xv[0]; xr[1] = xv[1]; xr[2] = xv[2]; xr[3] = xv[3];
}
__device__ __forceinline__ void A_to_lds(_Float16* __restrict__ adst, const v4f xr[4]) {
    v8h a0, a1;
#pragma unroll
    for (int i = 0; i < 8; ++i) {
        a0[i] = (_Float16)xr[i >> 2][i & 3];
        a1[i] = (_Float16)xr[2 + (i >> 2)][i & 3];
    }
    *(v8h*)(adst)     = a0;
    *(v8h*)(adst + 8) = a1;
}
// ---------------- fallback stage W: global i8 -> regs -> f16 LDS -----------------------
__device__ __forceinline__ void B_to_lds(_Float16* __restrict__ bdst, v16c wr) {
    v8h b0, b1;
#pragma unroll
    for (int i = 0; i < 8; ++i) {
        b0[i] = (_Float16)(float)wr[i];
        b1[i] = (_Float16)(float)wr[8 + i];
    }
    *(v8h*)(bdst)     = b0;
    *(v8h*)(bdst + 8) = b1;
}

// ---------------- main GEMM: 128x128 tile, 8 waves, f16 WMMA ---------------------------
// PREW=true : W already f16 in wv -> B tiles DMA'd global->LDS (async, no VALU, no ds_store)
// PREW=false: W is raw int8 in wv -> B tiles staged through regs with conversion
template <bool PREW>
__global__ __launch_bounds__(256)
void ternary_linear_wmma(const float* __restrict__ x,
                         const void* __restrict__ wv,
                         const float* __restrict__ scale_p,
                         const float* __restrict__ bias,
                         float* __restrict__ out) {
    __shared__ __align__(16) _Float16 Ash[2][BM * LDSS];
    __shared__ __align__(16) _Float16 Bsh[2][BN * LDSS];

    const int tid  = threadIdx.x;
    const int lane = tid & 31;
    const int wave = tid >> 5;
    const int l16  = lane & 15;
    const int g    = lane >> 4;

    const int    nBase = blockIdx.x * BN;
    const size_t mBase = (size_t)blockIdx.y * BM;

    const int wm = wave >> 2;            // 0..1 : 64 M-rows each
    const int wn = wave & 3;             // 0..3 : 32 N-cols each

    const int lrow   = tid >> 1;         // 0..127
    const int lhalf  = tid & 1;          // 0..1
    const int ldsOff = lrow * LDSS + lhalf * 16;

    const float*       xp   = x + (mBase + (size_t)lrow) * D_IN + lhalf * 16;
    const _Float16*    wp16 = (const _Float16*)wv    + (size_t)(nBase + lrow) * D_IN + lhalf * 16;
    const signed char* wp8  = (const signed char*)wv + (size_t)(nBase + lrow) * D_IN + lhalf * 16;

    v8f acc[4][2] = {};

    // ---- preload tile 0 ----
    v4f  xr[4];
    v16c wr;
    stageA_regs(xp, 0, xr);
    A_to_lds(&Ash[0][ldsOff], xr);
    if (PREW) {
        async_cp_b128(wp16,     &Bsh[0][ldsOff]);
        async_cp_b128(wp16 + 8, &Bsh[0][ldsOff + 8]);
        wait_async0();
    } else {
        wr = *(const v16c*)(wp8);
        B_to_lds(&Bsh[0][ldsOff], wr);
    }
    __syncthreads();

    for (int kt = 0; kt < KTILES; ++kt) {
        const int  buf  = kt & 1;
        const bool more = (kt + 1 < KTILES);
        const int  kn   = (kt + 1) * BK;

        if (more) {
            stageA_regs(xp, kn, xr);
            if (PREW) {   // kick off B DMA for next buffer; lands during the WMMA block
                async_cp_b128(wp16 + kn,     &Bsh[buf ^ 1][ldsOff]);
                async_cp_b128(wp16 + kn + 8, &Bsh[buf ^ 1][ldsOff + 8]);
            } else {
                wr = *(const v16c*)(wp8 + kn);
            }
        }

        const _Float16* Awave = &Ash[buf][(wm * 64) * LDSS];
        const _Float16* Bwave = &Bsh[buf][(wn * 32) * LDSS];

        v16h af[4];
#pragma unroll
        for (int mt = 0; mt < 4; ++mt) {
            const _Float16* p = Awave + (mt * 16 + l16) * LDSS + 8 * g;
            v8h lo = *(const v8h*)p;
            v8h hi = *(const v8h*)(p + 16);
            af[mt] = __builtin_shufflevector(lo, hi, 0, 1, 2, 3, 4, 5, 6, 7,
                                             8, 9, 10, 11, 12, 13, 14, 15);
        }
        v16h bf[2];
#pragma unroll
        for (int nt = 0; nt < 2; ++nt) {
            const _Float16* p = Bwave + (nt * 16 + l16) * LDSS + 8 * g;
            v8h lo = *(const v8h*)p;
            v8h hi = *(const v8h*)(p + 16);
            bf[nt] = __builtin_shufflevector(lo, hi, 0, 1, 2, 3, 4, 5, 6, 7,
                                             8, 9, 10, 11, 12, 13, 14, 15);
        }

#pragma unroll
        for (int mt = 0; mt < 4; ++mt)
#pragma unroll
            for (int nt = 0; nt < 2; ++nt)
                acc[mt][nt] = __builtin_amdgcn_wmma_f32_16x16x32_f16(
                    false, af[mt], false, bf[nt],
                    (short)0, acc[mt][nt], false, false);

        if (more) {
            A_to_lds(&Ash[buf ^ 1][ldsOff], xr);
            if (!PREW) B_to_lds(&Bsh[buf ^ 1][ldsOff], wr);
        }
        if (PREW) wait_async0();   // own async B copies complete; barrier publishes them
        __syncthreads();
    }

    // ---- epilogue: y = acc * scale + bias ----
    const float s = scale_p[0];
#pragma unroll
    for (int nt = 0; nt < 2; ++nt) {
        const int   n  = nBase + wn * 32 + nt * 16 + l16;
        const float bv = bias[n];
#pragma unroll
        for (int mt = 0; mt < 4; ++mt) {
            const size_t m  = mBase + (size_t)(wm * 64 + mt * 16 + 8 * g);
            float*       op = out + m * D_OUT + n;
#pragma unroll
            for (int r = 0; r < 8; ++r)
                op[(size_t)r * D_OUT] = acc[mt][nt][r] * s + bv;
        }
    }
}

extern "C" void kernel_launch(void* const* d_in, const int* in_sizes, int n_in,
                              void* d_out, int out_size, void* d_ws, size_t ws_size,
                              hipStream_t stream) {
    const float*       x     = (const float*)d_in[0];
    const signed char* w     = (const signed char*)d_in[1];
    const float*       scale = (const float*)d_in[2];
    const float*       bias  = (const float*)d_in[3];
    float*             out   = (float*)d_out;

    dim3 grid(D_OUT / BN, M_TOT / BM);   // (86, 64)

    if (ws_size >= WS_NEED) {
        // one-time dequant of ternary W to f16 (exact), then async-DMA GEMM
        const int cvt_blocks = (D_OUT * D_IN) / (256 * 16);   // 11008
        w_to_f16<<<cvt_blocks, 256, 0, stream>>>(w, (_Float16*)d_ws);
        ternary_linear_wmma<true><<<grid, 256, 0, stream>>>(x, d_ws, scale, bias, out);
    } else {
        ternary_linear_wmma<false><<<grid, 256, 0, stream>>>(x, w, scale, bias, out);
    }
}
